// EncoderLayerWithLatentRelations_22170621182362
// MI455X (gfx1250) — matrix operand order repaired and checked by reading, not compiled, not run on hardware
//
#include <hip/hip_runtime.h>

typedef __attribute__((ext_vector_type(16))) _Float16 v16h;
typedef __attribute__((ext_vector_type(8)))  float    v8f;
typedef __attribute__((ext_vector_type(8)))  _Float16 h8;

#define Q_ 256
#define C_ 128
#define T_ 128
#define L_ 512
#define D_ 1024
#define H_ 8
#define DK_ 128
#define FF_ 4096

// ---------------------------------------------------------------------------
// WMMA GEMM: C[M,N] = A[M,K](f16) * B(f16) (+bias)(+relu)(+residual)
//   TRANSB=1: B is [N,K] row-major (weight layout, y = x @ W.T)
//   TRANSB=0: B is [K,N] row-major
//   GUARD=0 : requires M%128==0, N%128==0, K%32==0 (no bounds code at all)
// 256 threads = 8 waves; block tile 128x128; each wave owns a 32x64 sub-tile
// (2 A-frags x 4 B-frags -> 8 v_wmma per K-step). Batched via grid.z.
// ---------------------------------------------------------------------------
template<bool TRANSB, bool GUARD>
__global__ __launch_bounds__(256)
void gemm_wmma_kernel(const _Float16* __restrict__ A, long sAb, int lda,
                      const _Float16* __restrict__ B, long sBb, int ldb,
                      float* __restrict__ Cf, _Float16* __restrict__ Ch,
                      long sCb, int ldc,
                      const float* __restrict__ bias,
                      const float* __restrict__ resid, long sRb,
                      int M, int N, int K, int relu)
{
    __shared__ _Float16 lsA[128][40];   // 32 K + 8 pad (keeps 16B alignment)
    __shared__ _Float16 lsB[128][40];   // stored [n][k]

    const int b   = blockIdx.z;
    A += (long)b * sAb;
    B += (long)b * sBb;
    const int n0  = blockIdx.x * 128;
    const int m0  = blockIdx.y * 128;
    const int t   = threadIdx.x;
    const int wv  = t >> 5;
    const int ln  = t & 31;
    const int wm  = wv >> 1;           // 0..3 : 32-row strip
    const int wn  = wv & 1;            // 0..1 : 64-col strip
    const int row = ln & 15;
    const int kh8 = (ln >> 4) << 3;    // A k-offset: 0 or 8
    const int kb16= (ln >> 4) << 4;    // unused: B frags use kh8 too

    v8f acc[2][4];
    #pragma unroll
    for (int mi = 0; mi < 2; ++mi)
        #pragma unroll
        for (int ns = 0; ns < 4; ++ns)
            #pragma unroll
            for (int e = 0; e < 8; ++e) acc[mi][ns][e] = 0.0f;

    for (int k0 = 0; k0 < K; k0 += 32) {
        // ---- stage A tile [128 x 32] ----
        {
            int r  = t >> 1;
            int cb = (t & 1) * 16;
            #pragma unroll
            for (int s = 0; s < 2; ++s) {
                int kk = cb + s * 8;
                int gm = m0 + r, gk = k0 + kk;
                if (!GUARD || (gm < M && gk + 8 <= K)) {
                    *(h8*)&lsA[r][kk] = *(const h8*)&A[(long)gm * lda + gk];
                } else {
                    #pragma unroll
                    for (int e = 0; e < 8; ++e) {
                        int ke = gk + e;
                        lsA[r][kk + e] = (gm < M && ke < K)
                                         ? A[(long)gm * lda + ke] : (_Float16)0.0f;
                    }
                }
            }
        }
        // ---- stage B tile -> lsB[n][k] ----
        if (TRANSB) {
            int r  = t >> 1;
            int cb = (t & 1) * 16;
            #pragma unroll
            for (int s = 0; s < 2; ++s) {
                int kk = cb + s * 8;
                int gn = n0 + r, gk = k0 + kk;
                if (!GUARD || (gn < N && gk + 8 <= K)) {
                    *(h8*)&lsB[r][kk] = *(const h8*)&B[(long)gn * ldb + gk];
                } else {
                    #pragma unroll
                    for (int e = 0; e < 8; ++e) {
                        int ke = gk + e;
                        lsB[r][kk + e] = (gn < N && ke < K)
                                         ? B[(long)gn * ldb + ke] : (_Float16)0.0f;
                    }
                }
            }
        } else {
            int n  = t & 127;
            int kb = (t >> 7) * 16;
            #pragma unroll
            for (int s = 0; s < 16; ++s) {
                int kk = kb + s;
                int gk = k0 + kk, gn = n0 + n;
                if (!GUARD || (gk < K && gn < N))
                    lsB[n][kk] = B[(long)gk * ldb + gn];
                else
                    lsB[n][kk] = (_Float16)0.0f;
            }
        }
        __syncthreads();

        // ---- A fragments (16-bit A 16x32 layout per ISA table) ----
        v16h af[2];
        #pragma unroll
        for (int mi = 0; mi < 2; ++mi) {
            const int arow = wm * 32 + mi * 16 + row;
            #pragma unroll
            for (int r2 = 0; r2 < 4; ++r2) {
                af[mi][2*r2]         = lsA[arow][kh8 + 2*r2];
                af[mi][2*r2 + 1]     = lsA[arow][kh8 + 2*r2 + 1];
                af[mi][8 + 2*r2]     = lsA[arow][16 + kh8 + 2*r2];
                af[mi][8 + 2*r2 + 1] = lsA[arow][16 + kh8 + 2*r2 + 1];
            }
        }
        // ---- 4 B fragments x 2 A fragments = 8 WMMAs ----
        #pragma unroll
        for (int ns = 0; ns < 4; ++ns) {
            v16h bf;
            const int bcol = wn * 64 + ns * 16 + row;
            #pragma unroll
            for (int e = 0; e < 16; e += 2) {
                const int kb = (e < 8) ? (kh8 + e) : (8 + kh8 + e);
                bf[e]     = lsB[bcol][kb];
                bf[e + 1] = lsB[bcol][kb + 1];
            }
            acc[0][ns] = __builtin_amdgcn_wmma_f32_16x16x32_f16(
                false, af[0], false, bf, (short)0, acc[0][ns], false, false);
            acc[1][ns] = __builtin_amdgcn_wmma_f32_16x16x32_f16(
                false, af[1], false, bf, (short)0, acc[1][ns], false, false);
        }
        __syncthreads();
    }

    // ---- epilogue: C layout = VGPR r -> row r + 8*(lane>=16), col = lane&15 ----
    #pragma unroll
    for (int mi = 0; mi < 2; ++mi) {
        const int rbase = m0 + wm * 32 + mi * 16 + ((ln >> 4) << 3);
        #pragma unroll
        for (int ns = 0; ns < 4; ++ns) {
            int col = n0 + wn * 64 + ns * 16 + (ln & 15);
            #pragma unroll
            for (int r2 = 0; r2 < 8; ++r2) {
                int rr = rbase + r2;
                if (!GUARD || (rr < M && col < N)) {
                    float v = acc[mi][ns][r2];
                    if (bias)  v += bias[col];
                    if (relu)  v = v > 0.0f ? v : 0.0f;
                    if (resid) v += resid[(long)b * sRb + (long)rr * ldc + col];
                    long o = (long)b * sCb + (long)rr * ldc + col;
                    if (Cf) Cf[o] = v;
                    if (Ch) Ch[o] = (_Float16)v;
                }
            }
        }
    }
}

// ---------------------------------------------------------------------------
__global__ void f32_to_f16_kernel(const float* __restrict__ in,
                                  _Float16* __restrict__ out, long n)
{
    long i = (long)blockIdx.x * blockDim.x + threadIdx.x;
    if (i < n) out[i] = (_Float16)in[i];
}

// one block (256 thr) per row; D = 1024
__global__ __launch_bounds__(256)
void layernorm_kernel(const float* __restrict__ x, const float* __restrict__ g,
                      const float* __restrict__ bta, _Float16* __restrict__ out)
{
    int rowi = blockIdx.x;
    const float* xr = x + (long)rowi * D_;
    __shared__ float red[8];
    float s = 0.0f;
    for (int c = threadIdx.x; c < D_; c += 256) s += xr[c];
    #pragma unroll
    for (int o = 16; o > 0; o >>= 1) s += __shfl_xor(s, o, 32);
    if ((threadIdx.x & 31) == 0) red[threadIdx.x >> 5] = s;
    __syncthreads();
    float tot = 0.0f;
    #pragma unroll
    for (int w = 0; w < 8; ++w) tot += red[w];
    float mean = tot * (1.0f / D_);
    __syncthreads();
    float v = 0.0f;
    for (int c = threadIdx.x; c < D_; c += 256) {
        float d0 = xr[c] - mean; v += d0 * d0;
    }
    #pragma unroll
    for (int o = 16; o > 0; o >>= 1) v += __shfl_xor(v, o, 32);
    if ((threadIdx.x & 31) == 0) red[threadIdx.x >> 5] = v;
    __syncthreads();
    float vt = 0.0f;
    #pragma unroll
    for (int w = 0; w < 8; ++w) vt += red[w];
    float rstd = rsqrtf(vt * (1.0f / D_) + 1e-5f);
    for (int c = threadIdx.x; c < D_; c += 256)
        out[(long)rowi * D_ + c] = (_Float16)((xr[c] - mean) * rstd * g[c] + bta[c]);
}

// u[h,i, 4types x 3f] = q[h,i,:] . l*_k[f,:]   grid (512, 8), block 128
__global__ __launch_bounds__(128)
void uproj_kernel(const _Float16* __restrict__ qh,
                  const float* __restrict__ lqc, const float* __restrict__ lqt,
                  const float* __restrict__ lcq, const float* __restrict__ ltq,
                  float* __restrict__ u)
{
    int i = blockIdx.x, h = blockIdx.y, d = threadIdx.x;
    float qd = (float)qh[(long)i * D_ + h * DK_ + d];
    __shared__ float red[4];
    const float* tabs[4] = {lqc, lqt, lcq, ltq};
    for (int tb = 0; tb < 4; ++tb)
        for (int f = 0; f < 3; ++f) {
            float p = qd * tabs[tb][f * DK_ + d];
            #pragma unroll
            for (int o = 16; o > 0; o >>= 1) p += __shfl_xor(p, o, 32);
            __syncthreads();
            if ((d & 31) == 0) red[d >> 5] = p;
            __syncthreads();
            if (d == 0)
                u[((long)h * L_ + i) * 12 + tb * 3 + f]
                    = red[0] + red[1] + red[2] + red[3];
            __syncthreads();
        }
}

// scores += relation bias, scale, mask.   grid (512, 8), block 256
__global__ __launch_bounds__(256)
void score_bias_kernel(float* __restrict__ scores,
                       const float* __restrict__ tk,   // [8,512,64]
                       const float* __restrict__ u,    // [8,512,12]
                       const int* __restrict__ pre, const int* __restrict__ ctr,
                       const float* __restrict__ qc, const float* __restrict__ cq,
                       const float* __restrict__ qt, const float* __restrict__ tq,
                       const int* __restrict__ mask)
{
    int i = blockIdx.x, h = blockIdx.y;
    __shared__ float tloc[50];
    __shared__ float uloc[12];
    if (threadIdx.x < 50) tloc[threadIdx.x] = tk[((long)h * L_ + i) * 64 + threadIdx.x];
    if (threadIdx.x < 12) uloc[threadIdx.x] = u[((long)h * L_ + i) * 12 + threadIdx.x];
    __syncthreads();
    const float scale = 0.08838834764831845f;  // 1/sqrt(128)
    float* srow = scores + ((long)h * L_ + i) * L_;
    for (int j = threadIdx.x; j < L_; j += 256) {
        float sc = srow[j] + tloc[pre[i * L_ + j]];
        if (i < Q_) {
            if (j < Q_) sc += tloc[0];
            else if (j < Q_ + C_) {
                const float* f3 = qc + ((long)i * C_ + (j - Q_)) * 3;
                sc += f3[0]*uloc[0] + f3[1]*uloc[1] + f3[2]*uloc[2];
            } else {
                const float* f3 = qt + ((long)i * T_ + (j - Q_ - C_)) * 3;
                sc += f3[0]*uloc[3] + f3[1]*uloc[4] + f3[2]*uloc[5];
            }
        } else {
            if (j < Q_) {
                if (i < Q_ + C_) {
                    const float* f3 = cq + ((long)(i - Q_) * Q_ + j) * 3;
                    sc += f3[0]*uloc[6] + f3[1]*uloc[7] + f3[2]*uloc[8];
                } else {
                    const float* f3 = tq + ((long)(i - Q_ - C_) * Q_ + j) * 3;
                    sc += f3[0]*uloc[9] + f3[1]*uloc[10] + f3[2]*uloc[11];
                }
            } else {
                sc += tloc[ctr[(i - Q_) * (C_ + T_) + (j - Q_)]];
            }
        }
        sc *= scale;
        if (mask[i * L_ + j] == 0) sc = -1e9f;
        srow[j] = sc;
    }
}

// row softmax, 512 wide.  grid (512, 8), block 256
__global__ __launch_bounds__(256)
void softmax_kernel(const float* __restrict__ scores, _Float16* __restrict__ p)
{
    int i = blockIdx.x, h = blockIdx.y;
    const float* srow = scores + ((long)h * L_ + i) * L_;
    _Float16* prow = p + ((long)h * L_ + i) * L_;
    float a = srow[threadIdx.x], b = srow[threadIdx.x + 256];
    float m = fmaxf(a, b);
    __shared__ float red[8];
    #pragma unroll
    for (int o = 16; o > 0; o >>= 1) m = fmaxf(m, __shfl_xor(m, o, 32));
    if ((threadIdx.x & 31) == 0) red[threadIdx.x >> 5] = m;
    __syncthreads();
    float mm = red[0];
    #pragma unroll
    for (int w = 1; w < 8; ++w) mm = fmaxf(mm, red[w]);
    float ea = __expf(a - mm), eb = __expf(b - mm);
    float s = ea + eb;
    __syncthreads();
    #pragma unroll
    for (int o = 16; o > 0; o >>= 1) s += __shfl_xor(s, o, 32);
    if ((threadIdx.x & 31) == 0) red[threadIdx.x >> 5] = s;
    __syncthreads();
    float tot = 0.0f;
    #pragma unroll
    for (int w = 0; w < 8; ++w) tot += red[w];
    float inv = 1.0f / tot;
    prow[threadIdx.x]       = (_Float16)(ea * inv);
    prow[threadIdx.x + 256] = (_Float16)(eb * inv);
}

// bin-sum p by relation id + latent 3-vec sums.  grid (512, 8), block 256
__global__ __launch_bounds__(256)
void pv_bins_kernel(const _Float16* __restrict__ p,
                    const int* __restrict__ pre, const int* __restrict__ ctr,
                    const float* __restrict__ qc, const float* __restrict__ cq,
                    const float* __restrict__ qt, const float* __restrict__ tq,
                    _Float16* __restrict__ sbins,  // [8,512,64]
                    float* __restrict__ w)         // [8,512,6]
{
    int i = blockIdx.x, h = blockIdx.y;
    __shared__ float bins[64];
    __shared__ float wloc[6];
    if (threadIdx.x < 64) bins[threadIdx.x] = 0.0f;
    if (threadIdx.x < 6)  wloc[threadIdx.x] = 0.0f;
    __syncthreads();
    const _Float16* prow = p + ((long)h * L_ + i) * L_;
    float w0=0,w1=0,w2=0,w3=0,w4=0,w5=0;
    for (int j = threadIdx.x; j < L_; j += 256) {
        float pv = (float)prow[j];
        atomicAdd(&bins[pre[i * L_ + j]], pv);
        if (i < Q_) {
            if (j < Q_) atomicAdd(&bins[0], pv);
            else if (j < Q_ + C_) {
                const float* f3 = qc + ((long)i * C_ + (j - Q_)) * 3;
                w0 += pv*f3[0]; w1 += pv*f3[1]; w2 += pv*f3[2];
            } else {
                const float* f3 = qt + ((long)i * T_ + (j - Q_ - C_)) * 3;
                w3 += pv*f3[0]; w4 += pv*f3[1]; w5 += pv*f3[2];
            }
        } else {
            if (j < Q_) {
                if (i < Q_ + C_) {
                    const float* f3 = cq + ((long)(i - Q_) * Q_ + j) * 3;
                    w0 += pv*f3[0]; w1 += pv*f3[1]; w2 += pv*f3[2];
                } else {
                    const float* f3 = tq + ((long)(i - Q_ - C_) * Q_ + j) * 3;
                    w3 += pv*f3[0]; w4 += pv*f3[1]; w5 += pv*f3[2];
                }
            } else {
                atomicAdd(&bins[ctr[(i - Q_) * (C_ + T_) + (j - Q_)]], pv);
            }
        }
    }
    atomicAdd(&wloc[0], w0); atomicAdd(&wloc[1], w1); atomicAdd(&wloc[2], w2);
    atomicAdd(&wloc[3], w3); atomicAdd(&wloc[4], w4); atomicAdd(&wloc[5], w5);
    __syncthreads();
    if (threadIdx.x < 64)
        sbins[((long)h * L_ + i) * 64 + threadIdx.x] = (_Float16)bins[threadIdx.x];
    if (threadIdx.x < 6)
        w[((long)h * L_ + i) * 6 + threadIdx.x] = wloc[threadIdx.x];
}

// o32 + latent w contributions -> attn_h (f16, [512,1024])
__global__ __launch_bounds__(256)
void attn_combine_kernel(const float* __restrict__ o32, const float* __restrict__ w,
                         const float* __restrict__ lqcv, const float* __restrict__ lqtv,
                         const float* __restrict__ lcqv, const float* __restrict__ ltqv,
                         _Float16* __restrict__ attn)
{
    int idx = blockIdx.x * 256 + threadIdx.x;     // 512*1024
    int i = idx >> 10, c = idx & 1023;
    int h = c >> 7, d = c & 127;
    const float* wr = w + ((long)h * L_ + i) * 6;
    float v = o32[idx];
    if (i < Q_) {
        v += wr[0]*lqcv[d] + wr[1]*lqcv[DK_ + d] + wr[2]*lqcv[2*DK_ + d];
        v += wr[3]*lqtv[d] + wr[4]*lqtv[DK_ + d] + wr[5]*lqtv[2*DK_ + d];
    } else {
        v += wr[0]*lcqv[d] + wr[1]*lcqv[DK_ + d] + wr[2]*lcqv[2*DK_ + d];
        v += wr[3]*ltqv[d] + wr[4]*ltqv[DK_ + d] + wr[5]*ltqv[2*DK_ + d];
    }
    attn[idx] = (_Float16)v;
}

// ---------------------------------------------------------------------------
static inline void launch_gemm(hipStream_t st,
    const _Float16* A, long sAb, int lda,
    const _Float16* B, long sBb, int ldb, int transB,
    float* Cf, _Float16* Ch, long sCb, int ldc,
    const float* bias, const float* resid, long sRb,
    int M, int N, int K, int relu, int batch, int guard)
{
    dim3 grid((N + 127) / 128, (M + 127) / 128, batch);
    if (transB) {
        if (guard)
            gemm_wmma_kernel<true, true><<<grid, 256, 0, st>>>(A, sAb, lda, B, sBb, ldb,
                Cf, Ch, sCb, ldc, bias, resid, sRb, M, N, K, relu);
        else
            gemm_wmma_kernel<true, false><<<grid, 256, 0, st>>>(A, sAb, lda, B, sBb, ldb,
                Cf, Ch, sCb, ldc, bias, resid, sRb, M, N, K, relu);
    } else {
        if (guard)
            gemm_wmma_kernel<false, true><<<grid, 256, 0, st>>>(A, sAb, lda, B, sBb, ldb,
                Cf, Ch, sCb, ldc, bias, resid, sRb, M, N, K, relu);
        else
            gemm_wmma_kernel<false, false><<<grid, 256, 0, st>>>(A, sAb, lda, B, sBb, ldb,
                Cf, Ch, sCb, ldc, bias, resid, sRb, M, N, K, relu);
    }
}

static inline void launch_cvt(hipStream_t st, const float* in, _Float16* out, long n)
{
    f32_to_f16_kernel<<<(int)((n + 255) / 256), 256, 0, st>>>(in, out, n);
}

extern "C" void kernel_launch(void* const* d_in, const int* in_sizes, int n_in,
                              void* d_out, int out_size, void* d_ws, size_t ws_size,
                              hipStream_t stream)
{
    const float* x    = (const float*)d_in[0];
    const float* qcR  = (const float*)d_in[1];
    const float* cqR  = (const float*)d_in[2];
    const float* qtR  = (const float*)d_in[3];
    const float* tqR  = (const float*)d_in[4];
    const int*   ctR  = (const int*)d_in[5];
    const int*   pre  = (const int*)d_in[6];
    const int*   mask = (const int*)d_in[7];
    const float* embk = (const float*)d_in[8];
    const float* embv = (const float*)d_in[9];
    const float* lqck = (const float*)d_in[10]; const float* lqcv = (const float*)d_in[11];
    const float* lcqk = (const float*)d_in[12]; const float* lcqv = (const float*)d_in[13];
    const float* lqtk = (const float*)d_in[14]; const float* lqtv = (const float*)d_in[15];
    const float* ltqk = (const float*)d_in[16]; const float* ltqv = (const float*)d_in[17];
    const float* Wq = (const float*)d_in[18]; const float* bq = (const float*)d_in[19];
    const float* Wk = (const float*)d_in[20]; const float* bk = (const float*)d_in[21];
    const float* Wv = (const float*)d_in[22]; const float* bv = (const float*)d_in[23];
    const float* Wo = (const float*)d_in[24]; const float* bo = (const float*)d_in[25];
    const float* ln1g = (const float*)d_in[26]; const float* ln1b = (const float*)d_in[27];
    const float* ln2g = (const float*)d_in[28]; const float* ln2b = (const float*)d_in[29];
    const float* W1 = (const float*)d_in[30]; const float* b1 = (const float*)d_in[31];
    const float* W2 = (const float*)d_in[32]; const float* b2 = (const float*)d_in[33];
    (void)in_sizes; (void)n_in; (void)out_size; (void)ws_size;

    char* wsp = (char*)d_ws;
    auto alloc = [&](size_t bytes) -> char* {
        char* p = wsp; wsp += (bytes + 255) & ~(size_t)255; return p;
    };
    _Float16* Wq_h = (_Float16*)alloc((size_t)D_*D_*2);
    _Float16* Wk_h = (_Float16*)alloc((size_t)D_*D_*2);
    _Float16* Wv_h = (_Float16*)alloc((size_t)D_*D_*2);
    _Float16* Wo_h = (_Float16*)alloc((size_t)D_*D_*2);
    _Float16* W1_h = (_Float16*)alloc((size_t)FF_*D_*2);
    _Float16* W2_h = (_Float16*)alloc((size_t)D_*FF_*2);
    _Float16* ek_h = (_Float16*)alloc((size_t)50*DK_*2);
    _Float16* ev_h = (_Float16*)alloc((size_t)50*DK_*2);
    _Float16* xln_h = (_Float16*)alloc((size_t)L_*D_*2);
    _Float16* q_h  = (_Float16*)alloc((size_t)L_*D_*2);
    _Float16* k_h  = (_Float16*)alloc((size_t)L_*D_*2);
    _Float16* v_h  = (_Float16*)alloc((size_t)L_*D_*2);
    float*    sco  = (float*)   alloc((size_t)H_*L_*L_*4);
    _Float16* p_h  = (_Float16*)alloc((size_t)H_*L_*L_*2);
    float*    tk   = (float*)   alloc((size_t)H_*L_*64*4);
    float*    ubuf = (float*)   alloc((size_t)H_*L_*12*4);
    _Float16* sb_h = (_Float16*)alloc((size_t)H_*L_*64*2);
    float*    wbuf = (float*)   alloc((size_t)H_*L_*6*4);
    float*    o32  = (float*)   alloc((size_t)L_*D_*4);
    _Float16* at_h = (_Float16*)alloc((size_t)L_*D_*2);
    float*    hbuf = (float*)   alloc((size_t)L_*D_*4);
    _Float16* ln2h = (_Float16*)alloc((size_t)L_*D_*2);
    _Float16* ff_h = (_Float16*)alloc((size_t)L_*FF_*2);

    // ---- weight / table conversion to f16 ----
    launch_cvt(stream, Wq, Wq_h, (long)D_*D_);
    launch_cvt(stream, Wk, Wk_h, (long)D_*D_);
    launch_cvt(stream, Wv, Wv_h, (long)D_*D_);
    launch_cvt(stream, Wo, Wo_h, (long)D_*D_);
    launch_cvt(stream, W1, W1_h, (long)FF_*D_);
    launch_cvt(stream, W2, W2_h, (long)D_*FF_);
    launch_cvt(stream, embk, ek_h, 50L*DK_);
    launch_cvt(stream, embv, ev_h, 50L*DK_);

    // ---- LN1 ----
    layernorm_kernel<<<L_, 256, 0, stream>>>(x, ln1g, ln1b, xln_h);

    // ---- QKV projections (WMMA, unguarded fast path) ----
    launch_gemm(stream, xln_h, 0, D_, Wq_h, 0, D_, 1, nullptr, q_h, 0, D_,
                bq, nullptr, 0, L_, D_, D_, 0, 1, 0);
    launch_gemm(stream, xln_h, 0, D_, Wk_h, 0, D_, 1, nullptr, k_h, 0, D_,
                bk, nullptr, 0, L_, D_, D_, 0, 1, 0);
    launch_gemm(stream, xln_h, 0, D_, Wv_h, 0, D_, 1, nullptr, v_h, 0, D_,
                bv, nullptr, 0, L_, D_, D_, 0, 1, 0);

    // ---- scores = q @ k^T, per head (WMMA, batched over heads) ----
    launch_gemm(stream, q_h, DK_, D_, k_h, DK_, D_, 1, sco, nullptr,
                (long)L_*L_, L_, nullptr, nullptr, 0, L_, L_, DK_, 0, H_, 0);

    // ---- t[h,i,r] = q . emb_k[r] (WMMA, ragged N=50 -> guarded) ----
    launch_gemm(stream, q_h, DK_, D_, ek_h, 0, DK_, 1, tk, nullptr,
                (long)L_*64, 64, nullptr, nullptr, 0, L_, 50, DK_, 0, H_, 1);

    // ---- u[h,i,f] small projections ----
    uproj_kernel<<<dim3(L_, H_), 128, 0, stream>>>(q_h, lqck, lqtk, lcqk, ltqk, ubuf);

    // ---- relation bias + scale + mask ----
    score_bias_kernel<<<dim3(L_, H_), 256, 0, stream>>>(
        sco, tk, ubuf, pre, ctR, qcR, cqR, qtR, tqR, mask);

    // ---- softmax ----
    softmax_kernel<<<dim3(L_, H_), 256, 0, stream>>>(sco, p_h);

    // ---- o = p @ v, per head (WMMA), written directly in [i, h*128+d] ----
    launch_gemm(stream, p_h, (long)L_*L_, L_, v_h, DK_, D_, 0, o32, nullptr,
                DK_, D_, nullptr, nullptr, 0, L_, DK_, L_, 0, H_, 0);

    // ---- bin sums of p + latent w sums ----
    pv_bins_kernel<<<dim3(L_, H_), 256, 0, stream>>>(
        p_h, pre, ctR, qcR, cqR, qtR, tqR, sb_h, wbuf);

    // ---- o += s @ emb_v (WMMA, ragged K=50 -> guarded, residual accumulate) ----
    launch_gemm(stream, sb_h, (long)L_*64, 64, ev_h, 0, DK_, 0, o32, nullptr,
                DK_, D_, nullptr, o32, DK_, L_, DK_, 50, 0, H_, 1);

    // ---- add latent-v contributions, pack to f16 ----
    attn_combine_kernel<<<(L_*D_)/256, 256, 0, stream>>>(
        o32, wbuf, lqcv, lqtv, lcqv, ltqv, at_h);

    // ---- output projection + residual x  -> h ----
    launch_gemm(stream, at_h, 0, D_, Wo_h, 0, D_, 1, hbuf, nullptr, 0, D_,
                bo, x, 0, L_, D_, D_, 0, 1, 0);

    // ---- LN2 ----
    layernorm_kernel<<<L_, 256, 0, stream>>>(hbuf, ln2g, ln2b, ln2h);

    // ---- FFN1 (relu) ----
    launch_gemm(stream, ln2h, 0, D_, W1_h, 0, D_, 1, nullptr, ff_h, 0, FF_,
                b1, nullptr, 0, L_, FF_, D_, 1, 1, 0);

    // ---- FFN2 + residual h -> d_out ----
    launch_gemm(stream, ff_h, 0, FF_, W2_h, 0, FF_, 1, (float*)d_out, nullptr,
                0, D_, b2, hbuf, 0, L_, D_, FF_, 0, 1, 0);
}
